// FASA_90340342104549
// MI455X (gfx1250) — compile-verified, matrix operand order checked by reading, not run on hardware
//
#include <hip/hip_runtime.h>

// ---------------------------------------------------------------------------
// FASA block for MI455X (gfx1250, wave32, WMMA).
// All GEMM-like math (1x1 convs + attention) uses v_wmma_f32_16x16x32_bf16.
// Fragment layouts follow CDNA5 ISA:
//   16-bit A 16x32 : lane l -> row m=l&15 ; halves j<8 -> K=(l>>4)*8+j,
//                    j>=8 -> K=16+(l>>4)*8+(j-8)
//   16-bit B 32x16 : symmetric (lane l -> col n=l&15, same K mapping)
//   f32 C/D 16x16  : lane l -> col n=l&15 ; VGPR r -> row m=r+8*(l>>4)
// ---------------------------------------------------------------------------

#define BB    16
#define DIM   256
#define HH    56
#define WW    56
#define HW    (HH * WW)      // 3136
#define HEADS 8
#define DH    32
#define S1    (28 * 28)      // 784
#define S2    (14 * 14)      // 196
#define KPAD  224            // keys padded to 7*32 for WMMA K-chunks

typedef __attribute__((ext_vector_type(16))) __bf16 v16bf;
typedef __attribute__((ext_vector_type(8)))  __bf16 v8bf;
typedef __attribute__((ext_vector_type(8)))  float  v8f;

static __device__ __forceinline__ v8f wmma_bf16(v16bf a, v16bf b, v8f c) {
  return __builtin_amdgcn_wmma_f32_16x16x32_bf16(
      /*neg_a=*/false, a, /*neg_b=*/false, b,
      /*c_mod=*/(short)0, c, /*reuse_a=*/false, /*reuse_b=*/false);
}

// Two contiguous 16B chunks -> one 16x32 (or 32x16) bf16 fragment per lane.
static __device__ __forceinline__ v16bf frag16(const __bf16* lo, const __bf16* hi) {
  v8bf l = *(const v8bf*)lo;
  v8bf h = *(const v8bf*)hi;
  v16bf r;
#pragma unroll
  for (int i = 0; i < 8; ++i) { r[i] = l[i]; r[i + 8] = h[i]; }
  return r;
}

// ---------------------------------------------------------------------------
// Elementwise fp32 -> bf16 (weights)
// ---------------------------------------------------------------------------
__global__ void f32_to_bf16_kernel(const float* __restrict__ in,
                                   __bf16* __restrict__ out, int n) {
  int i = blockIdx.x * blockDim.x + threadIdx.x;
  if (i < n) out[i] = (__bf16)in[i];
}

// ---------------------------------------------------------------------------
// (b, C, S) fp32  ->  (b, S, C) bf16   (activation staging for GEMM B-frags)
// ---------------------------------------------------------------------------
__global__ void f32_to_bf16_t_kernel(const float* __restrict__ in,
                                     __bf16* __restrict__ out, int C, int S) {
  long long idx = (long long)blockIdx.x * blockDim.x + threadIdx.x;
  long long total = (long long)BB * C * S;
  if (idx >= total) return;
  int s = (int)(idx % S);
  long long t = idx / S;
  int c = (int)(t % C);
  int b = (int)(t / C);
  out[((size_t)b * S + s) * C + c] = (__bf16)in[idx];
}

// ---------------------------------------------------------------------------
// Generic WMMA GEMM: Out[b][o][s] = sum_c W[o][c] * Xt[b][s][c] + bias[o]
// One wave computes a 16(M=o) x 32(N=s) tile; K-loop over C in steps of 32.
// ---------------------------------------------------------------------------
__global__ __launch_bounds__(32)
void gemm_bf16_kernel(const __bf16* __restrict__ W, const __bf16* __restrict__ Xt,
                      const float* __restrict__ bias, float* __restrict__ Out,
                      int C, int O, int S) {
  const int lane = threadIdx.x & 31;
  const int n    = lane & 15;
  const int koff = (lane >> 4) * 8;
  const int s0 = blockIdx.x * 32;
  const int o0 = blockIdx.y * 16;
  const int b  = blockIdx.z;

  const __bf16* arow = W + (size_t)(o0 + n) * C + koff;   // A row m = lane&15
  int sa = s0 + n;      if (sa > S - 1) sa = S - 1;       // clamp: pad cols
  int sb = s0 + 16 + n; if (sb > S - 1) sb = S - 1;
  const __bf16* brow0 = Xt + ((size_t)b * S + sa) * C + koff;
  const __bf16* brow1 = Xt + ((size_t)b * S + sb) * C + koff;

  v8f acc0 = {}, acc1 = {};
#pragma unroll 4
  for (int k = 0; k < C; k += 32) {
    v16bf a  = frag16(arow  + k, arow  + k + 16);
    v16bf b0 = frag16(brow0 + k, brow0 + k + 16);
    v16bf b1 = frag16(brow1 + k, brow1 + k + 16);
    acc0 = wmma_bf16(a, b0, acc0);
    acc1 = wmma_bf16(a, b1, acc1);
  }

#pragma unroll
  for (int r = 0; r < 8; ++r) {
    int o = o0 + r + 8 * (lane >> 4);
    float bv = bias[o];
    size_t base = ((size_t)b * O + o) * S;
    int s_lo = s0 + n;
    int s_hi = s0 + 16 + n;
    if (s_lo < S) Out[base + s_lo] = acc0[r] + bv;
    if (s_hi < S) Out[base + s_hi] = acc1[r] + bv;
  }
}

// ---------------------------------------------------------------------------
// Depthwise 5x5 stride-2 pad-2 conv + BatchNorm, NCHW fp32.
// ---------------------------------------------------------------------------
__global__ void dwconv5_bn_s2_kernel(const float* __restrict__ in,
                                     const float* __restrict__ w5,
                                     const float* __restrict__ bias,
                                     const float* __restrict__ g,
                                     const float* __restrict__ bb,
                                     const float* __restrict__ mean,
                                     const float* __restrict__ var,
                                     float* __restrict__ out,
                                     int Hi, int Wi, int Ho, int Wo) {
  int idx = blockIdx.x * blockDim.x + threadIdx.x;
  int total = BB * DIM * Ho * Wo;
  if (idx >= total) return;
  int xo = idx % Wo; int t = idx / Wo;
  int yo = t % Ho;   t /= Ho;
  int c  = t % DIM;  int b = t / DIM;
  const float* ip = in + ((size_t)b * DIM + c) * Hi * Wi;
  const float* wp = w5 + c * 25;
  float acc = bias[c];
#pragma unroll
  for (int dy = 0; dy < 5; ++dy) {
    int iy = yo * 2 - 2 + dy;
    if (iy < 0 || iy >= Hi) continue;
#pragma unroll
    for (int dx = 0; dx < 5; ++dx) {
      int ix = xo * 2 - 2 + dx;
      if (ix < 0 || ix >= Wi) continue;
      acc += ip[iy * Wi + ix] * wp[dy * 5 + dx];
    }
  }
  float scale = g[c] * rsqrtf(var[c] + 1e-5f);
  out[idx] = (acc - mean[c]) * scale + bb[c];
}

// ---------------------------------------------------------------------------
// Split kv (b,512,196) fp32 into K (b,h,key,dh) bf16 and V-transposed
// (b,h,dh,key) bf16, zero-padding keys to 224.
// ---------------------------------------------------------------------------
__global__ void kv_split_kernel(const float* __restrict__ kv,
                                __bf16* __restrict__ kt,
                                __bf16* __restrict__ vt) {
  int idx = blockIdx.x * blockDim.x + threadIdx.x;
  int total = BB * HEADS * KPAD * DH;
  if (idx >= total) return;
  int dh  = idx % DH;   int t = idx / DH;
  int key = t % KPAD;   t /= KPAD;
  int h   = t % HEADS;  int b = t / HEADS;
  float kval = 0.f, vval = 0.f;
  if (key < S2) {
    size_t base = ((size_t)b * (2 * DIM) + h * DH + dh) * S2 + key;
    kval = kv[base];
    vval = kv[base + (size_t)DIM * S2];
  }
  kt[(((size_t)(b * HEADS + h)) * KPAD + key) * DH + dh] = (__bf16)kval;
  vt[(((size_t)(b * HEADS + h)) * DH + dh) * KPAD + key] = (__bf16)vval;
}

// ---------------------------------------------------------------------------
// Attention: one wave per (b, head, 16 Q rows). DH=32 == one WMMA K-span.
//   S = scale * Q K^T  (13 key tiles) -> LDS -> softmax -> P (bf16, LDS)
//   O = P V            (7 K-chunks x 2 N-tiles), normalized on store.
// ---------------------------------------------------------------------------
__global__ __launch_bounds__(32)
void attn_kernel(const __bf16* __restrict__ qt,   // (b, q, 256) bf16
                 const __bf16* __restrict__ kt,   // (b, h, 224, 32) bf16
                 const __bf16* __restrict__ vt,   // (b, h, 32, 224) bf16
                 float* __restrict__ gfeat) {     // (b, 256, 3136) fp32
  __shared__ __align__(16) float  sS[16 * KPAD];
  __shared__ __align__(16) __bf16 sP[16 * KPAD];
  __shared__ float sSum[16];

  const int lane = threadIdx.x & 31;
  const int m    = lane & 15;
  const int koff = (lane >> 4) * 8;
  const int b  = blockIdx.z;
  const int h  = blockIdx.y;
  const int q0 = blockIdx.x * 16;
  const float scal = 0.17677669529663687f;        // DH^-0.5

  // Q fragment (A): row m -> q0+m, K -> dh
  const __bf16* qrow = qt + ((size_t)b * HW + q0 + m) * DIM + h * DH + koff;
  v16bf aq = frag16(qrow, qrow + 16);

  const __bf16* kbase = kt + ((size_t)(b * HEADS + h)) * KPAD * DH;
#pragma unroll
  for (int t = 0; t < 13; ++t) {                  // keys 0..207 (rows>=196 are 0)
    const __bf16* krow = kbase + (size_t)(t * 16 + m) * DH + koff;
    v16bf bk = frag16(krow, krow + 16);
    v8f z = {};
    v8f s = wmma_bf16(aq, bk, z);
#pragma unroll
    for (int r = 0; r < 8; ++r)
      sS[(r + 8 * (lane >> 4)) * KPAD + t * 16 + m] = s[r] * scal;
  }
  __syncthreads();

  if (lane < 16) {                                // lane == Q row
    float mx = -1e30f;
    for (int j = 0; j < S2; ++j) mx = fmaxf(mx, sS[lane * KPAD + j]);
    float sum = 0.f;
    for (int j = 0; j < S2; ++j) {
      float e = __expf(sS[lane * KPAD + j] - mx);
      sum += e;
      sP[lane * KPAD + j] = (__bf16)e;            // unnormalized P
    }
    for (int j = S2; j < KPAD; ++j) sP[lane * KPAD + j] = (__bf16)0.f;
    sSum[lane] = sum;
  }
  __syncthreads();

  v8f o0 = {}, o1 = {};
  const __bf16* vbase = vt + ((size_t)(b * HEADS + h)) * DH * KPAD;
#pragma unroll
  for (int t = 0; t < 7; ++t) {                   // key chunks of 32
    const __bf16* prow = &sP[m * KPAD + t * 32 + koff];
    v16bf ap = frag16(prow, prow + 16);
    const __bf16* v0 = vbase + (size_t)m * KPAD + t * 32 + koff;        // dh=n
    const __bf16* v1 = vbase + (size_t)(16 + m) * KPAD + t * 32 + koff; // dh=16+n
    o0 = wmma_bf16(ap, frag16(v0, v0 + 16), o0);
    o1 = wmma_bf16(ap, frag16(v1, v1 + 16), o1);
  }

#pragma unroll
  for (int r = 0; r < 8; ++r) {
    int mr = r + 8 * (lane >> 4);
    int q  = q0 + mr;
    float inv = 1.f / sSum[mr];
    gfeat[((size_t)b * DIM + h * DH + m) * HW + q]      = o0[r] * inv;
    gfeat[((size_t)b * DIM + h * DH + 16 + m) * HW + q] = o1[r] * inv;
  }
}

// ---------------------------------------------------------------------------
// local mixer + gating, fused; writes transposed bf16 (b, s, c) so the
// final mix 1x1 conv can consume it directly as a WMMA B operand.
//   lf = dw5x5(q_local)+b ; lf *= sigmoid(lf) ; out = lf*sigmoid(g)*g
// ---------------------------------------------------------------------------
__global__ void fuse_kernel(const float* __restrict__ qloc,
                            const float* __restrict__ w5,
                            const float* __restrict__ bias,
                            const float* __restrict__ gfeat,
                            __bf16* __restrict__ fusedt) {
  int idx = blockIdx.x * blockDim.x + threadIdx.x;
  int total = BB * DIM * HW;
  if (idx >= total) return;
  int x = idx % WW; int t = idx / WW;
  int y = t % HH;   t /= HH;
  int c = t % DIM;  int b = t / DIM;
  const float* ip = qloc + ((size_t)b * DIM + c) * HW;
  const float* wp = w5 + c * 25;
  float acc = bias[c];
#pragma unroll
  for (int dy = 0; dy < 5; ++dy) {
    int iy = y - 2 + dy;
    if (iy < 0 || iy >= HH) continue;
#pragma unroll
    for (int dx = 0; dx < 5; ++dx) {
      int ix = x - 2 + dx;
      if (ix < 0 || ix >= WW) continue;
      acc += ip[iy * WW + ix] * wp[dy * 5 + dx];
    }
  }
  float lf = acc / (1.f + __expf(-acc));          // acc * sigmoid(acc)
  float gv = gfeat[idx];
  float val = lf * (1.f / (1.f + __expf(-gv))) * gv;
  fusedt[((size_t)b * HW + (y * WW + x)) * DIM + c] = (__bf16)val;
}

// ---------------------------------------------------------------------------
// Host orchestration
// ---------------------------------------------------------------------------
extern "C" void kernel_launch(void* const* d_in, const int* in_sizes, int n_in,
                              void* d_out, int out_size, void* d_ws, size_t ws_size,
                              hipStream_t stream) {
  (void)in_sizes; (void)n_in; (void)out_size; (void)ws_size;
  const float* x     = (const float*)d_in[0];
  const float* q_w   = (const float*)d_in[1];
  const float* q_b   = (const float*)d_in[2];
  const float* kv_w  = (const float*)d_in[3];
  const float* kv_b  = (const float*)d_in[4];
  const float* p0_w  = (const float*)d_in[5];
  const float* p0_b  = (const float*)d_in[6];
  const float* bn0_g = (const float*)d_in[7];
  const float* bn0_b = (const float*)d_in[8];
  const float* bn0_m = (const float*)d_in[9];
  const float* bn0_v = (const float*)d_in[10];
  const float* pl0_w = (const float*)d_in[11];
  const float* pl0_b = (const float*)d_in[12];
  const float* p1_w  = (const float*)d_in[13];
  const float* p1_b  = (const float*)d_in[14];
  const float* bn1_g = (const float*)d_in[15];
  const float* bn1_b = (const float*)d_in[16];
  const float* bn1_m = (const float*)d_in[17];
  const float* bn1_v = (const float*)d_in[18];
  const float* loc_w = (const float*)d_in[19];
  const float* loc_b = (const float*)d_in[20];
  const float* mix_w = (const float*)d_in[21];
  const float* mix_b = (const float*)d_in[22];
  float* out = (float*)d_out;

  char* ws = (char*)d_ws;
  auto alloc = [&](size_t bytes) -> char* {
    char* p = ws;
    ws += (bytes + 255) & ~(size_t)255;
    return p;
  };

  // bf16 weights
  __bf16* q_wb   = (__bf16*)alloc((size_t)DIM * DIM * 2);
  __bf16* kv_wb  = (__bf16*)alloc((size_t)2 * DIM * DIM * 2);
  __bf16* pl0_wb = (__bf16*)alloc((size_t)DIM * DIM * 2);
  __bf16* mix_wb = (__bf16*)alloc((size_t)DIM * DIM * 2);
  // region A: shared by xt -> qt -> fusedt (each lifetime disjoint)
  __bf16* regA   = (__bf16*)alloc((size_t)BB * HW * DIM * 2);
  float*  q_loc  = (float*)alloc((size_t)BB * DIM * HW * 4);
  float*  p0     = (float*)alloc((size_t)BB * DIM * S1 * 4);
  __bf16* p0t    = (__bf16*)alloc((size_t)BB * S1 * DIM * 2);
  float*  pmid   = (float*)alloc((size_t)BB * DIM * S1 * 4);
  float*  p1     = (float*)alloc((size_t)BB * DIM * S2 * 4);
  __bf16* p1t    = (__bf16*)alloc((size_t)BB * S2 * DIM * 2);
  float*  kvbuf  = (float*)alloc((size_t)BB * 2 * DIM * S2 * 4);
  __bf16* ktb    = (__bf16*)alloc((size_t)BB * HEADS * KPAD * DH * 2);
  __bf16* vtb    = (__bf16*)alloc((size_t)BB * HEADS * DH * KPAD * 2);
  float*  gfeat  = (float*)alloc((size_t)BB * DIM * HW * 4);

  const int TB = 256;
  auto blocks = [](long long n, int tb) { return (unsigned)((n + tb - 1) / tb); };

  // 1. weights -> bf16
  f32_to_bf16_kernel<<<blocks(DIM * DIM, TB), TB, 0, stream>>>(q_w, q_wb, DIM * DIM);
  f32_to_bf16_kernel<<<blocks(2 * DIM * DIM, TB), TB, 0, stream>>>(kv_w, kv_wb, 2 * DIM * DIM);
  f32_to_bf16_kernel<<<blocks(DIM * DIM, TB), TB, 0, stream>>>(pl0_w, pl0_wb, DIM * DIM);
  f32_to_bf16_kernel<<<blocks(DIM * DIM, TB), TB, 0, stream>>>(mix_w, mix_wb, DIM * DIM);

  // 2. x -> xt (regA), then q = 1x1 conv (WMMA GEMM)
  f32_to_bf16_t_kernel<<<blocks((long long)BB * DIM * HW, TB), TB, 0, stream>>>(x, regA, DIM, HW);
  gemm_bf16_kernel<<<dim3((HW + 31) / 32, DIM / 16, BB), 32, 0, stream>>>(
      q_wb, regA, q_b, q_loc, DIM, DIM, HW);

  // 3. pooled path: dw5 s2 + BN -> 1x1 (WMMA) -> dw5 s2 + BN -> kv 1x1 (WMMA)
  dwconv5_bn_s2_kernel<<<blocks((long long)BB * DIM * S1, TB), TB, 0, stream>>>(
      x, p0_w, p0_b, bn0_g, bn0_b, bn0_m, bn0_v, p0, HH, WW, 28, 28);
  f32_to_bf16_t_kernel<<<blocks((long long)BB * DIM * S1, TB), TB, 0, stream>>>(p0, p0t, DIM, S1);
  gemm_bf16_kernel<<<dim3((S1 + 31) / 32, DIM / 16, BB), 32, 0, stream>>>(
      pl0_wb, p0t, pl0_b, pmid, DIM, DIM, S1);
  dwconv5_bn_s2_kernel<<<blocks((long long)BB * DIM * S2, TB), TB, 0, stream>>>(
      pmid, p1_w, p1_b, bn1_g, bn1_b, bn1_m, bn1_v, p1, 28, 28, 14, 14);
  f32_to_bf16_t_kernel<<<blocks((long long)BB * DIM * S2, TB), TB, 0, stream>>>(p1, p1t, DIM, S2);
  gemm_bf16_kernel<<<dim3((S2 + 31) / 32, (2 * DIM) / 16, BB), 32, 0, stream>>>(
      kv_wb, p1t, kv_b, kvbuf, DIM, 2 * DIM, S2);
  kv_split_kernel<<<blocks((long long)BB * HEADS * KPAD * DH, TB), TB, 0, stream>>>(
      kvbuf, ktb, vtb);

  // 4. q_local -> qt (reuse regA; xt no longer needed), attention (WMMA)
  f32_to_bf16_t_kernel<<<blocks((long long)BB * DIM * HW, TB), TB, 0, stream>>>(
      q_loc, regA, DIM, HW);
  attn_kernel<<<dim3(HW / 16, HEADS, BB), 32, 0, stream>>>(regA, ktb, vtb, gfeat);

  // 5. local mixer + gating -> fusedt (reuse regA), then mix 1x1 (WMMA) -> out
  fuse_kernel<<<blocks((long long)BB * DIM * HW, TB), TB, 0, stream>>>(
      q_loc, loc_w, loc_b, gfeat, regA);
  gemm_bf16_kernel<<<dim3((HW + 31) / 32, DIM / 16, BB), 32, 0, stream>>>(
      mix_wb, regA, mix_b, out, DIM, DIM, HW);
}
